// MemoryUnit_81879256531350
// MI455X (gfx1250) — matrix operand order repaired and checked by reading, not compiled, not run on hardware
//
#include <hip/hip_runtime.h>

// ---------------------------------------------------------------------------
// MemoryUnit fused kernel for MI455X (gfx1250, wave32, WMMA bf16).
//   att   = softmax(clip(x) @ W^T / T)  -> hard_shrink_relu -> renormalize
//   out   = att @ W
// Outputs (concatenated in d_out): out [65536,256] fp32, att [65536,2000] fp32.
//
// Roofline: 134 GFLOP (2 GEMMs) vs ~640 MB mandatory HBM traffic (~27 us at
// 23.3 TB/s). bf16 WMMA keeps the matrix work in the same ballpark as the
// memory floor; fp32 softmax entirely in the 320KB-per-WGP LDS avoids a
// ~1 GB logits spill to HBM.
// ---------------------------------------------------------------------------

#define NROWS   65536
#define FEA     256
#define MEM     2000
#define MTILES  125          // MEM / 16
#define K1STEPS 8            // FEA / 32
#define K2STEPS 63           // ceil(MEM/32) -> padded K = 2016
#define KPAD    2016
#define LAMBDA  0.0025f
#define EPSV    1e-6f
#define INV_M   (1.0f / 2000.0f)

// LDS strides (elements), padded to avoid bank conflicts
#define LOG_STRIDE 2052      // fp32 logits row stride
#define ATT_STRIDE 2024      // bf16 att row stride
#define XA_STRIDE  264       // bf16 x row stride

#define SM_LOGITS_BYTES (16 * LOG_STRIDE * 4)            // 131328
#define SM_ATT_BYTES    (16 * ATT_STRIDE * 2)            // 64768
#define SM_XA_BYTES     (16 * XA_STRIDE * 2)             // 8448
#define SM_RED_BYTES    (16 * 16 * 4)                    // 1024
#define SMEM_BYTES (SM_LOGITS_BYTES + SM_ATT_BYTES + SM_XA_BYTES + SM_RED_BYTES)

typedef __attribute__((ext_vector_type(16))) __bf16 v16bf;
typedef __attribute__((ext_vector_type(8)))  __bf16 v8bf;
typedef __attribute__((ext_vector_type(8)))  float  v8f;

// Pack W into B-fragment order for V_WMMA_F32_16X16X32_BF16.
// B (32x16, KxN) per-lane layout (ISA 7.12.2): N = lane%16,
// K = (e/8)*16 + (lane/16)*8 + e%8 (matches A-fragment k order).

// GEMM1: B = W^T  (B[k=f][n=m] = W[m][f]).  Index: [tile(125)][ks(8)][lane(32)][e(16)]
__global__ void pack_w1(const float* __restrict__ W, __bf16* __restrict__ p) {
  int idx = blockIdx.x * 256 + threadIdx.x;              // 125*8*512 = 512000
  int e    = idx & 15;
  int lane = (idx >> 4) & 31;
  int ks   = (idx >> 9) % K1STEPS;
  int tile = (idx >> 9) / K1STEPS;
  int f = ks * 32 + ((e >> 3) << 4) + ((lane >> 4) << 3) + (e & 7);
  int m = tile * 16 + (lane & 15);
  p[idx] = (__bf16)W[m * FEA + f];
}

// GEMM2: B = W (B[k=m][n=f] = W[m][f]).  Index: [ftile(16)][ks(63)][lane(32)][e(16)]
__global__ void pack_w2(const float* __restrict__ W, __bf16* __restrict__ p) {
  int idx = blockIdx.x * 256 + threadIdx.x;              // 16*63*512 = 516096
  int e    = idx & 15;
  int lane = (idx >> 4) & 31;
  int ks   = (idx >> 9) % K2STEPS;
  int ft   = (idx >> 9) / K2STEPS;
  int f = ft * 16 + (lane & 15);
  int m = ks * 32 + ((e >> 3) << 4) + ((lane >> 4) << 3) + (e & 7);
  p[idx] = (m < MEM) ? (__bf16)W[m * FEA + f] : (__bf16)0.0f;
}

// A-fragment (16x32, MxK): lane row M = lane%16; two contiguous 8-elem runs:
// K = kb..kb+7 and 16+kb..16+kb+7, kb = (lane<16) ? 0 : 8.
__device__ __forceinline__ v16bf load_frag_A(const __bf16* row, int k0, int kb) {
  union { v16bf v; v8bf h[2]; } u;
  u.h[0] = *(const v8bf*)(row + k0 + kb);
  u.h[1] = *(const v8bf*)(row + k0 + 16 + kb);
  return u.v;
}

__device__ __forceinline__ float row_reduce(float v, int r, int sub, float* red, bool is_max) {
  red[r * 16 + sub] = v;
  __syncthreads();
  #pragma unroll
  for (int off = 8; off > 0; off >>= 1) {
    if (sub < off) {
      float a = red[r * 16 + sub], b = red[r * 16 + sub + off];
      red[r * 16 + sub] = is_max ? fmaxf(a, b) : (a + b);
    }
    __syncthreads();
  }
  float res = red[r * 16];
  __syncthreads();
  return res;
}

extern __shared__ char smem_raw[];

__global__ __launch_bounds__(256, 1)
void memunit_fused(const float* __restrict__ x, const float* __restrict__ temp,
                   const __bf16* __restrict__ packB1, const __bf16* __restrict__ packB2,
                   float* __restrict__ out, float* __restrict__ att_out) {
  float*  logits = (float*)smem_raw;                                     // [16][LOG_STRIDE]
  __bf16* attbf  = (__bf16*)(smem_raw + SM_LOGITS_BYTES);                // [16][ATT_STRIDE]
  __bf16* xa     = (__bf16*)(smem_raw + SM_LOGITS_BYTES + SM_ATT_BYTES); // [16][XA_STRIDE]
  float*  red    = (float*)(smem_raw + SM_LOGITS_BYTES + SM_ATT_BYTES + SM_XA_BYTES);

  const int tid  = threadIdx.x;
  const int wave = tid >> 5;
  const int lane = tid & 31;
  const int row0 = blockIdx.x * 16;
  const float invT = 1.0f / temp[0];

  // ---- Stage x block (clip -> bf16) into LDS -------------------------------
  #pragma unroll
  for (int i = 0; i < 16; ++i) {
    int idx = i * 256 + tid;              // 16 rows x 256 cols
    int r = idx >> 8, c = idx & 255;
    float v = x[(size_t)(row0 + r) * FEA + c];
    v = fminf(fmaxf(v, -10.0f), 10.0f);
    xa[r * XA_STRIDE + c] = (__bf16)v;
  }
  __syncthreads();

  // ---- GEMM1: logits[16 x 2000] = xa @ W^T  (bf16 WMMA, fp32 acc) ----------
  {
    const int m  = lane & 15;
    const int kb = (lane & 16) ? 8 : 0;
    const __bf16* arow = xa + m * XA_STRIDE;
    v16bf afr[K1STEPS];
    #pragma unroll
    for (int s = 0; s < K1STEPS; ++s) afr[s] = load_frag_A(arow, s * 32, kb);

    const int drow = ((lane >> 4) << 3);
    for (int tile = wave; tile < MTILES; tile += 8) {
      v8f acc = {};
      #pragma unroll
      for (int s = 0; s < K1STEPS; ++s) {
        v16bf b = *(const v16bf*)(packB1 + ((size_t)(tile * K1STEPS + s) * 32 + lane) * 16);
        acc = __builtin_amdgcn_wmma_f32_16x16x32_bf16(false, afr[s], false, b,
                                                      (short)0, acc, false, false);
      }
      // D layout: VGPR r -> row (lane<16 ? r : r+8), col lane%16
      int dcol = tile * 16 + (lane & 15);
      #pragma unroll
      for (int r = 0; r < 8; ++r) logits[(drow + r) * LOG_STRIDE + dcol] = acc[r];
    }
  }
  __syncthreads();

  // ---- Softmax(/T) + hard_shrink_relu + renormalize (all in LDS) -----------
  {
    const int r = tid >> 4, sub = tid & 15;
    float* lrow = logits + r * LOG_STRIDE;

    float mx = -3.0e38f;
    for (int m = sub; m < MEM; m += 16) mx = fmaxf(mx, lrow[m]);
    mx = row_reduce(mx, r, sub, red, true);

    float sm = 0.0f;
    for (int m = sub; m < MEM; m += 16) {
      float e = __expf((lrow[m] - mx) * invT);
      lrow[m] = e; sm += e;
    }
    sm = row_reduce(sm, r, sub, red, false);
    float inv_sum = 1.0f / sm;            // sm >= 1 (max term) -> finite

    float ws = 0.0f;
    for (int m = sub; m < MEM; m += 16) {
      float a = lrow[m] * inv_sum;
      a = fminf(fmaxf(a, -10.0f), 10.0f);
      float d = fmaxf(fabsf(a - LAMBDA) + EPSV, EPSV);
      float h = fmaxf(a - LAMBDA, 0.0f) * a / d;
      if (!(h == h) || fabsf(h) > 3.0e38f) h = 0.0f;
      lrow[m] = h; ws += h;
    }
    ws = row_reduce(ws, r, sub, red, false);
    const bool  zero  = ws < 1e-8f;
    const float scale = 1.0f / (ws + 1e-8f);

    for (int m = sub; m < MEM; m += 16) {
      float a = zero ? INV_M : lrow[m] * scale;
      if (!(a == a) || fabsf(a) > 3.0e38f) a = INV_M;
      att_out[(size_t)(row0 + r) * MEM + m] = a;   // final att output (fp32)
      attbf[r * ATT_STRIDE + m] = (__bf16)a;       // bf16 operand for GEMM2
    }
    attbf[r * ATT_STRIDE + 2000 + sub] = (__bf16)0.0f;  // zero-pad K to 2016
  }
  __syncthreads();

  // ---- GEMM2: out[16 x 256] = att @ W  (K = 2016, bf16 WMMA) ---------------
  // Two f-tiles per wave fused: shared A-fragment, two independent
  // accumulators -> back-to-back independent WMMAs each k-step.
  {
    const int m  = lane & 15;
    const int kb = (lane & 16) ? 8 : 0;
    const __bf16* arow = attbf + m * ATT_STRIDE;
    const int ft0 = wave * 2;             // 8 waves x 2 = 16 f-tiles
    const int ft1 = ft0 + 1;

    v8f acc0 = {}, acc1 = {};
    for (int s = 0; s < K2STEPS; ++s) {
      v16bf a  = load_frag_A(arow, s * 32, kb);
      v16bf b0 = *(const v16bf*)(packB2 + ((size_t)(ft0 * K2STEPS + s) * 32 + lane) * 16);
      v16bf b1 = *(const v16bf*)(packB2 + ((size_t)(ft1 * K2STEPS + s) * 32 + lane) * 16);
      acc0 = __builtin_amdgcn_wmma_f32_16x16x32_bf16(false, a, false, b0,
                                                     (short)0, acc0, false, false);
      acc1 = __builtin_amdgcn_wmma_f32_16x16x32_bf16(false, a, false, b1,
                                                     (short)0, acc1, false, false);
    }
    const int drow = ((lane >> 4) << 3);
    const int n = lane & 15;
    #pragma unroll
    for (int r = 0; r < 8; ++r) {
      out[(size_t)(row0 + drow + r) * FEA + ft0 * 16 + n] = acc0[r];
      out[(size_t)(row0 + drow + r) * FEA + ft1 * 16 + n] = acc1[r];
    }
  }
}

extern "C" void kernel_launch(void* const* d_in, const int* in_sizes, int n_in,
                              void* d_out, int out_size, void* d_ws, size_t ws_size,
                              hipStream_t stream) {
  const float* x    = (const float*)d_in[0];   // [65536, 256]
  const float* W    = (const float*)d_in[1];   // [2000, 256]
  const float* temp = (const float*)d_in[2];   // scalar

  float* out = (float*)d_out;                       // [65536, 256]
  float* att = out + (size_t)NROWS * FEA;           // [65536, 2000]

  __bf16* p1 = (__bf16*)d_ws;                       // 512000 elems (1.0 MB)
  __bf16* p2 = p1 + (size_t)MTILES * K1STEPS * 32 * 16;  // 516096 elems (1.0 MB)

  pack_w1<<<2000, 256, 0, stream>>>(W, p1);
  pack_w2<<<2016, 256, 0, stream>>>(W, p2);

  hipFuncSetAttribute((const void*)memunit_fused,
                      hipFuncAttributeMaxDynamicSharedMemorySize, SMEM_BYTES);
  memunit_fused<<<NROWS / 16, 256, SMEM_BYTES, stream>>>(x, temp, p1, p2, out, att);
}
